// MXFP4Attention_50852412785249
// MI455X (gfx1250) — matrix operand order repaired and checked by reading, not compiled, not run on hardware
//
#include <hip/hip_runtime.h>

// ---------------------------------------------------------------------------
// MXFP4 attention for MI455X (gfx1250, wave32, WMMA).
// v3 changes vs v2:
//  - __launch_bounds__(128, 1) on the WMMA kernels: the attention kernel
//    needs ~250 live VGPRs for resident K/V fragments; the default occupancy
//    heuristic collapsed them into one staging range and serialized
//    load->wait(0)->wmma. With the raised budget the 16 fragment loads are
//    issued ahead of the 8 score WMMAs and waits become partial.
// Everything else as v2: pre-swizzled bf16 weight tiles, pipelined GEMM,
// 1/sqrt(d) folded into Q, LDS tile transpose for V, global_prefetch_b8.
// Workspace: 128 MiB of d_ws.
// ---------------------------------------------------------------------------

typedef __attribute__((ext_vector_type(16))) __bf16 v16bf;
typedef __attribute__((ext_vector_type(8)))  float  v8f;

struct U128 { unsigned int x, y, z, w; };
struct F128 { float x, y, z, w; };
union FragU { v16bf v; U128 u[2]; };

__device__ __forceinline__ unsigned short f2bf(float f) {
  union { float f; unsigned int u; } cv; cv.f = f;
  unsigned int r = cv.u + 0x7FFFu + ((cv.u >> 16) & 1u);  // round-to-nearest-even
  return (unsigned short)(r >> 16);
}

// B-operand fragment: lane holds 16 contiguous bf16 (K-run of 16).
__device__ __forceinline__ v16bf load16(const unsigned short* p) {
  FragU f;
  f.u[0] = *(const U128*)(p);
  f.u[1] = *(const U128*)(p + 8);
  return f.v;
}

// A-operand fragment from a row-major matrix: p = row_base + k0 + (lane>>4)*16.
// Layout: lane<16 holds K = {0..15}, lane>=16 holds K = {16..31}.
__device__ __forceinline__ v16bf load_a_rm(const unsigned short* p) {
  FragU f;
  f.u[0] = *(const U128*)(p);        // K-run +0
  f.u[1] = *(const U128*)(p + 8);    // K-run +8
  return f.v;
}

__device__ __forceinline__ v8f wmma_bf16(v16bf a, v16bf b, v8f c) {
  return __builtin_amdgcn_wmma_f32_16x16x32_bf16(false, a, false, b, (short)0,
                                                 c, false, false);
}

// ---------------------------------------------------------------------------
// fp32 -> bf16 (row-major copy of X), 128-bit loads / 64-bit stores.
// ---------------------------------------------------------------------------
__global__ void k_f32_to_bf16(const float* __restrict__ x,
                              unsigned short* __restrict__ o, size_t n) {
  size_t i = ((size_t)blockIdx.x * blockDim.x + threadIdx.x) * 4;
  if (i >= n) return;
  F128 f = *(const F128*)(x + i);
  unsigned short tmp[4] = {f2bf(f.x), f2bf(f.y), f2bf(f.z), f2bf(f.w)};
  *(unsigned long long*)(o + i) = *(unsigned long long*)tmp;
}

// ---------------------------------------------------------------------------
// MXFP4 dequant -> bf16, pre-swizzled into WMMA B-operand tiles.
// Tile (kt, nt) = 32(K) x 16(N); lane l covers n = nt*16+(l&15),
// k = kt*32 + (l>>4)*16 + j, j=0..15 -> 16 contiguous bf16 per lane in memory.
// MXFP4 block_size == 32 == K-tile, so exactly one scale per lane.
// ---------------------------------------------------------------------------
__global__ void k_dequant_btile(const float* __restrict__ codes,   // [D, N]
                                const float* __restrict__ scales,  // [D/32, N]
                                unsigned short* __restrict__ Bt,
                                int D, int N) {
  int gt   = blockIdx.x * blockDim.x + threadIdx.x;
  int lane = gt & 31;
  int tile = gt >> 5;
  int ntiles = N >> 4;
  if (tile >= (D >> 5) * ntiles) return;
  int kt = tile / ntiles;
  int nt = tile - kt * ntiles;
  int n  = nt * 16 + (lane & 15);
  int k0 = kt * 32 + (lane >> 4) * 16;
  float sc = scales[(size_t)kt * N + n];
  unsigned short tmp[16];
  #pragma unroll
  for (int j = 0; j < 16; ++j)
    tmp[j] = f2bf(codes[(size_t)(k0 + j) * N + n] * sc);
  unsigned short* out = Bt + (size_t)tile * 512 + lane * 16;
  *(U128*)out       = *(U128*)tmp;
  *(U128*)(out + 8) = *(U128*)(tmp + 8);
}

// ---------------------------------------------------------------------------
// WMMA GEMM: C[M,N] = ((A[M,K] * B) + bias) * oscale.
// Block = 128 threads = 4 waves (2x2), block tile 128x128, wave tile 64x64
// (4x4 accumulator fragments). No LDS: operands stream from the 192MB L2.
// Software-pipelined: next K-step fragments are loaded before the current
// step's 16 WMMAs, so loads hide under matrix ops (partial loadcnt waits).
// ---------------------------------------------------------------------------
template <int OUT_F32>
__global__ __launch_bounds__(128, 1)
void k_gemm_bf16(const unsigned short* __restrict__ A,
                 const unsigned short* __restrict__ Bt,
                 const float* __restrict__ bias,
                 void* __restrict__ Cout,
                 int M, int N, int K, float oscale) {
  const int lane = threadIdx.x & 31;
  const int wave = threadIdx.x >> 5;
  const int lq   = lane & 15;
  const int half = lane >> 4;
  const int m0 = blockIdx.y * 128 + (wave >> 1) * 64;
  const int n0 = blockIdx.x * 128 + (wave & 1) * 64;
  const int ntiles = N >> 4;

  v8f acc[4][4] = {};
  v16bf af[2][4], bf[2][4];

  const unsigned short* arow[4];
  #pragma unroll
  for (int i = 0; i < 4; ++i)
    arow[i] = A + (size_t)(m0 + i * 16 + lq) * K + half * 16;
  const unsigned short* btile0 = Bt + (size_t)((n0 >> 4)) * 512 + lane * 16;

  auto load_frags = [&](int buf, int k0) {
    #pragma unroll
    for (int i = 0; i < 4; ++i) af[buf][i] = load_a_rm(arow[i] + k0);
    const size_t koff = (size_t)(k0 >> 5) * ntiles * 512;
    #pragma unroll
    for (int j = 0; j < 4; ++j)
      bf[buf][j] = load16(btile0 + koff + (size_t)j * 512);
  };

  load_frags(0, 0);
  for (int k0 = 0; k0 < K; k0 += 32) {
    const int cur = (k0 >> 5) & 1;
    if (k0 + 32 < K) {
      load_frags(cur ^ 1, k0 + 32);
      if (k0 + 64 < K) {  // pull A rows 2 steps ahead (global_prefetch_b8)
        #pragma unroll
        for (int i = 0; i < 4; ++i)
          __builtin_prefetch(arow[i] + k0 + 64, 0, 3);
      }
    }
    #pragma unroll
    for (int i = 0; i < 4; ++i)
      #pragma unroll
      for (int j = 0; j < 4; ++j)
        acc[i][j] = wmma_bf16(af[cur][i], bf[cur][j], acc[i][j]);
  }

  // C/D layout: lane -> col n0+j*16+(lane&15); VGPR r -> row r + 8*(lane>>4).
  #pragma unroll
  for (int j = 0; j < 4; ++j) {
    const int col = n0 + j * 16 + lq;
    const float bv = bias[col];
    #pragma unroll
    for (int i = 0; i < 4; ++i) {
      #pragma unroll
      for (int r = 0; r < 8; ++r) {
        const int row = m0 + i * 16 + r + half * 8;
        float val = (acc[i][j][r] + bv) * oscale;
        if (OUT_F32) ((float*)Cout)[(size_t)row * N + col] = val;
        else ((unsigned short*)Cout)[(size_t)row * N + col] = f2bf(val);
      }
    }
  }
}

// ---------------------------------------------------------------------------
// V transpose: [B*S, H*Dh] -> [B, H, Dh, S], coalesced both sides via a
// padded 32x32 LDS tile. Grid: (S/32, H*(Dh/32), B), block 256 (= 32x8).
// ---------------------------------------------------------------------------
__global__ __launch_bounds__(256)
void k_transpose_v(const unsigned short* __restrict__ V,
                   unsigned short* __restrict__ Vt,
                   int S, int H, int Dh, int F) {
  __shared__ unsigned short tile[32][33];
  const int tx = threadIdx.x & 31;
  const int ty = threadIdx.x >> 5;           // 0..7
  const int s0 = blockIdx.x * 32;
  const int h  = blockIdx.y >> 2;            // Dh/32 == 4
  const int d0 = (blockIdx.y & 3) * 32;
  const int b  = blockIdx.z;

  #pragma unroll
  for (int i = 0; i < 4; ++i) {
    const int s = s0 + ty + i * 8;
    tile[ty + i * 8][tx] = V[(size_t)(b * S + s) * F + h * Dh + d0 + tx];
  }
  __syncthreads();
  #pragma unroll
  for (int i = 0; i < 4; ++i) {
    const int d = d0 + ty + i * 8;
    Vt[((size_t)((b * H + h) * Dh + d)) * S + s0 + tx] = tile[tx][ty + i * 8];
  }
}

// ---------------------------------------------------------------------------
// Flash attention: grid (S/64, H, B), 128 threads = 4 waves, each wave owns a
// 16-query tile. Per 32-key step: load all 8 K-frags and 8 V-frags first
// (V loads overlap score WMMAs + softmax VALU), 8 WMMAs for S = Q*K^T,
// online softmax with wave32 shuffles, P fp32(D-layout) -> bf16 A-layout via
// per-wave LDS transpose (DScnt-ordered), then 8 WMMAs for O += P*V.
// NOTE: 1/sqrt(d) is already folded into Q by the Q-projection GEMM.
// __launch_bounds__(128, 1): ~250 live VGPRs wanted (O 64, Q 32, K 64, V 64);
// without it the allocator collapses the fragment buffers and serializes.
// ---------------------------------------------------------------------------
__global__ __launch_bounds__(128, 1)
void k_attention(const unsigned short* __restrict__ Q,
                 const unsigned short* __restrict__ Kk,
                 const unsigned short* __restrict__ Vt,
                 unsigned short* __restrict__ O,
                 int S, int H, int Dh, int F) {
  __shared__ unsigned short pbuf[4 * 16 * 32];  // 1KB per wave
  const int lane = threadIdx.x & 31;
  const int wave = threadIdx.x >> 5;
  const int lq   = lane & 15;
  const int half = lane >> 4;
  const int b  = blockIdx.z;
  const int h  = blockIdx.y;
  const int q0 = blockIdx.x * 64 + wave * 16;
  unsigned short* myP = pbuf + wave * 512;

  // Q fragments (pre-scaled by 1/sqrt(d)): 4 chunks of K=32, loaded once.
  v16bf aq[4];
  const size_t rowQ = (size_t)(b * S + q0 + lq) * F + h * Dh;
  #pragma unroll
  for (int c = 0; c < 4; ++c)
    aq[c] = load_a_rm(Q + rowQ + c * 32 + half * 16);

  // Per-lane base pointers for K (this lane's key column) and V^T rows.
  const unsigned short* kbase =
      Kk + (size_t)(b * S + lq) * F + h * Dh + half * 16;        // + key*F + c4*32
  const unsigned short* vbase =
      Vt + ((size_t)((b * H + h) * Dh + lq)) * S + half * 16;    // + j*16*S + kb

  v8f o[8] = {};
  float run_m[8], run_l[8];
  #pragma unroll
  for (int r = 0; r < 8; ++r) { run_m[r] = -1e30f; run_l[r] = 0.0f; }

  for (int kb = 0; kb < S; kb += 32) {
    // ---- load all operands for this 32-key step up front ----------------
    v16bf bk[2][4];
    #pragma unroll
    for (int t = 0; t < 2; ++t) {
      const unsigned short* kp = kbase + (size_t)(kb + t * 16) * F;
      #pragma unroll
      for (int c4 = 0; c4 < 4; ++c4) bk[t][c4] = load16(kp + c4 * 32);
    }
    v16bf bv[8];
    #pragma unroll
    for (int j = 0; j < 8; ++j)
      bv[j] = load16(vbase + (size_t)j * 16 * S + kb);

    // ---- scores: two 16-key sub-tiles (waits only on K frags) -----------
    v8f s[2];
    #pragma unroll
    for (int t = 0; t < 2; ++t) {
      v8f c = {};
      #pragma unroll
      for (int c4 = 0; c4 < 4; ++c4) c = wmma_bf16(aq[c4], bk[t][c4], c);
      s[t] = c;
    }

    // ---- online softmax over 32 keys ------------------------------------
    float alpha[8], p0[8], p1[8];
    #pragma unroll
    for (int r = 0; r < 8; ++r) {
      float tm = fmaxf(s[0][r], s[1][r]);
      tm = fmaxf(tm, __shfl_xor(tm, 1, 32));
      tm = fmaxf(tm, __shfl_xor(tm, 2, 32));
      tm = fmaxf(tm, __shfl_xor(tm, 4, 32));
      tm = fmaxf(tm, __shfl_xor(tm, 8, 32));
      float nm = fmaxf(run_m[r], tm);
      alpha[r] = __expf(run_m[r] - nm);
      p0[r] = __expf(s[0][r] - nm);
      p1[r] = __expf(s[1][r] - nm);
      float sum = p0[r] + p1[r];
      sum += __shfl_xor(sum, 1, 32);
      sum += __shfl_xor(sum, 2, 32);
      sum += __shfl_xor(sum, 4, 32);
      sum += __shfl_xor(sum, 8, 32);
      run_l[r] = run_l[r] * alpha[r] + sum;
      run_m[r] = nm;
    }
    #pragma unroll
    for (int j = 0; j < 8; ++j)
      #pragma unroll
      for (int r = 0; r < 8; ++r)
        o[j][r] *= alpha[r];

    // ---- P (D-layout fp32) -> A-layout bf16 via LDS ---------------------
    #pragma unroll
    for (int r = 0; r < 8; ++r) {
      const int m = r + half * 8;
      myP[m * 32 + lq]      = f2bf(p0[r]);
      myP[m * 32 + 16 + lq] = f2bf(p1[r]);
    }
    v16bf pa = load_a_rm(myP + lq * 32 + half * 16);  // DScnt-ordered vs stores

    // ---- O += P * V (V frags already resident) --------------------------
    #pragma unroll
    for (int j = 0; j < 8; ++j) o[j] = wmma_bf16(pa, bv[j], o[j]);
  }

  // ---- normalize + store bf16 -------------------------------------------
  #pragma unroll
  for (int r = 0; r < 8; ++r) run_l[r] = 1.0f / run_l[r];
  #pragma unroll
  for (int j = 0; j < 8; ++j) {
    const int col = h * Dh + j * 16 + lq;
    #pragma unroll
    for (int r = 0; r < 8; ++r) {
      const int row = b * S + q0 + r + half * 8;
      O[(size_t)row * F + col] = f2bf(o[j][r] * run_l[r]);
    }
  }
}

// ---------------------------------------------------------------------------
extern "C" void kernel_launch(void* const* d_in, const int* in_sizes, int n_in,
                              void* d_out, int out_size, void* d_ws, size_t ws_size,
                              hipStream_t stream) {
  (void)in_sizes; (void)n_in; (void)out_size; (void)ws_size;
  const int Bb = 2, S = 2048, D = 2048, H = 16, Dh = 128;
  const int F = H * Dh;          // 2048
  const int M = Bb * S;          // 4096
  const float qscale = 0.08838834764831845f;  // 1/sqrt(Dh), folded into Q

  const float* x    = (const float*)d_in[0];
  const float* wq_c = (const float*)d_in[1];
  const float* wq_s = (const float*)d_in[2];
  const float* bq   = (const float*)d_in[3];
  const float* wk_c = (const float*)d_in[4];
  const float* wk_s = (const float*)d_in[5];
  const float* bk   = (const float*)d_in[6];
  const float* wv_c = (const float*)d_in[7];
  const float* wv_s = (const float*)d_in[8];
  const float* bv   = (const float*)d_in[9];
  const float* wo_c = (const float*)d_in[10];
  const float* wo_s = (const float*)d_in[11];
  const float* bo   = (const float*)d_in[12];

  // Workspace layout (128 MiB total)
  char* ws = (char*)d_ws;
  const size_t MB = 1024 * 1024;
  unsigned short* xb  = (unsigned short*)(ws);            // 16 MiB  X bf16
  unsigned short* wqb = (unsigned short*)(ws + 16 * MB);  // 8 MiB each, B-tiled
  unsigned short* wkb = (unsigned short*)(ws + 24 * MB);
  unsigned short* wvb = (unsigned short*)(ws + 32 * MB);
  unsigned short* wob = (unsigned short*)(ws + 40 * MB);
  unsigned short* qb  = (unsigned short*)(ws + 48 * MB);  // 16 MiB
  unsigned short* kbf = (unsigned short*)(ws + 64 * MB);  // 16 MiB
  unsigned short* vbf = (unsigned short*)(ws + 80 * MB);  // 16 MiB
  unsigned short* vt  = (unsigned short*)(ws + 96 * MB);  // 16 MiB
  unsigned short* ao  = (unsigned short*)(ws + 112 * MB); // 16 MiB

  // 1) X -> bf16
  {
    size_t n = (size_t)M * D;
    k_f32_to_bf16<<<(unsigned)((n / 4 + 255) / 256), 256, 0, stream>>>(x, xb, n);
  }
  // 2) Dequant all four weights into WMMA B-tiles
  {
    int threads = (D >> 5) * (F >> 4) * 32;
    unsigned g = (unsigned)((threads + 255) / 256);
    k_dequant_btile<<<g, 256, 0, stream>>>(wq_c, wq_s, wqb, D, F);
    k_dequant_btile<<<g, 256, 0, stream>>>(wk_c, wk_s, wkb, D, F);
    k_dequant_btile<<<g, 256, 0, stream>>>(wv_c, wv_s, wvb, D, F);
    k_dequant_btile<<<g, 256, 0, stream>>>(wo_c, wo_s, wob, F, D);
  }
  // 3) Q/K/V projections (bf16 out); Q pre-scaled by 1/sqrt(Dh)
  {
    dim3 gg(F / 128, M / 128);
    k_gemm_bf16<0><<<gg, 128, 0, stream>>>(xb, wqb, bq, qb, M, F, D, qscale);
    k_gemm_bf16<0><<<gg, 128, 0, stream>>>(xb, wkb, bk, kbf, M, F, D, 1.0f);
    k_gemm_bf16<0><<<gg, 128, 0, stream>>>(xb, wvb, bv, vbf, M, F, D, 1.0f);
  }
  // 4) V transpose (coalesced LDS tile transpose)
  {
    dim3 gt(S / 32, H * (Dh / 32), Bb);
    k_transpose_v<<<gt, 256, 0, stream>>>(vbf, vt, S, H, Dh, F);
  }
  // 5) Flash attention
  {
    dim3 ga(S / 64, H, Bb);
    k_attention<<<ga, 128, 0, stream>>>(qb, kbf, vt, ao, S, H, Dh, F);
  }
  // 6) Output projection (fp32 out to d_out, + bo)
  {
    dim3 go(D / 128, M / 128);
    k_gemm_bf16<1><<<go, 128, 0, stream>>>(ao, wob, bo, d_out, M, D, F, 1.0f);
  }
}
